// myGATConv_58420145160705
// MI455X (gfx1250) — matrix operand order repaired and checked by reading, not compiled, not run on hardware
//
#include <hip/hip_runtime.h>
#include <hip/hip_bf16.h>
#include <stdint.h>

#define N_NODES   30000
#define N_EDGES   480000
#define IN_F      256
#define N_HEADS   8
#define OUT_F     32
#define EF        64
#define N_ETYPES  8

typedef __attribute__((ext_vector_type(16))) __bf16 v16bf;
typedef __attribute__((ext_vector_type(8)))  float  v8f;

// ---------------------------------------------------------------------------
// init: zero rst (d_out part 1) and denom, set emax sentinel
// ---------------------------------------------------------------------------
__global__ void k_init(float* __restrict__ rst, float* __restrict__ denom,
                       int* __restrict__ emax, int nr, int nd) {
    int i = blockIdx.x * blockDim.x + threadIdx.x;
    if (i < nr) rst[i] = 0.0f;
    if (i < nd) { denom[i] = 0.0f; emax[i] = (int)0x80000000; }
}

// ---------------------------------------------------------------------------
// f32 -> bf16 (round to nearest even), stored as u16
// ---------------------------------------------------------------------------
__global__ void k_cvt_bf16(const float* __restrict__ in,
                           unsigned short* __restrict__ out, int n) {
    int i = blockIdx.x * blockDim.x + threadIdx.x;
    if (i >= n) return;
    unsigned int u = __float_as_uint(in[i]);
    u += 0x7fffu + ((u >> 16) & 1u);
    out[i] = (unsigned short)(u >> 16);
}

// ---------------------------------------------------------------------------
// per-edge-type table: e_emb -> MLP -> ee[type][head]  (only 8 types!)
// one block, 128 threads
// ---------------------------------------------------------------------------
__global__ void k_typeprep(const float* __restrict__ edge_emb,
                           const float* __restrict__ W1, const float* __restrict__ b1,
                           const float* __restrict__ W2, const float* __restrict__ b2,
                           const float* __restrict__ attn_e,
                           float* __restrict__ ee_t /*[8][8]*/) {
    __shared__ float h1[N_ETYPES][2 * EF];     // 8 x 128
    __shared__ float ee_s[N_ETYPES][N_HEADS];
    const int t = threadIdx.x;                 // 0..127
    // stage 1: h1 = relu(W1 @ e_emb + b1); thread t computes row t for all types
    for (int ty = 0; ty < N_ETYPES; ++ty) {
        float acc = b1[t];
        for (int k = 0; k < EF; ++k)
            acc += W1[t * EF + k] * edge_emb[ty * EF + k];
        h1[ty][t] = fmaxf(acc, 0.0f);
    }
    if (t < N_ETYPES * N_HEADS) ((float*)ee_s)[t] = 0.0f;
    __syncthreads();
    // stage 2: h2 = W2 @ h1 + b2 (512 rows), reduce against attn_e into ee
    for (int ty = 0; ty < N_ETYPES; ++ty) {
        for (int r = t; r < N_HEADS * EF; r += 128) {
            float acc = b2[r];
            for (int k = 0; k < 2 * EF; ++k)
                acc += W2[r * (2 * EF) + k] * h1[ty][k];
            int h = r / EF, f = r % EF;
            atomicAdd(&ee_s[ty][h], acc * attn_e[h * EF + f]);
        }
    }
    __syncthreads();
    if (t < N_ETYPES * N_HEADS) ((float*)ee_t)[t] = ((float*)ee_s)[t];
}

// ---------------------------------------------------------------------------
// WMMA GEMM: feat_src = feat(bf16) @ fc_W(bf16)^T, f32 accumulate
// block = 256 threads (8 waves); each wave computes one 16x16 tile, K=256.
// block tile = 32 rows x 64 cols. A/B fed in ISA 7.12.2 register layouts.
// ---------------------------------------------------------------------------
__global__ void k_gemm_wmma(const unsigned short* __restrict__ featb,
                            const unsigned short* __restrict__ fcwb,
                            float* __restrict__ feat_src) {
    const int lane = threadIdx.x & 31;
    const int wave = threadIdx.x >> 5;                 // 0..7
    const int row0 = (blockIdx.x * 2 + (wave & 1)) * 16;
    const int col0 = (blockIdx.y * 4 + (wave >> 1)) * 16;

    const int l15 = lane & 15;
    const int hi  = lane >> 4;                          // 0 or 1
    int arow = row0 + l15;
    if (arow >= N_NODES) arow = N_NODES - 1;            // clamp: OOB rows never stored
    const int akoff = hi * 8;                           // A: lanes16-31 hold K= +8..15 / +24..31
    const int bkoff = hi * 8;                           // B: lanes16-31 hold K= +8..15 / +24..31

    const unsigned short* aptr = featb + (size_t)arow * IN_F;
    const unsigned short* bptr = fcwb + (size_t)(col0 + l15) * IN_F; // B[k][n]=fc_W[n][k]

    v8f acc = {0.f, 0.f, 0.f, 0.f, 0.f, 0.f, 0.f, 0.f};
#pragma unroll
    for (int kb = 0; kb < IN_F; kb += 32) {
        union { v16bf bf; uint4 u[2]; } A, B;
        A.u[0] = *reinterpret_cast<const uint4*>(aptr + kb + akoff);       // K = kb+koff..+7
        A.u[1] = *reinterpret_cast<const uint4*>(aptr + kb + 16 + akoff);  // K = kb+16+koff..+7
        B.u[0] = *reinterpret_cast<const uint4*>(bptr + kb + bkoff);       // K = kb+bk..+7
        B.u[1] = *reinterpret_cast<const uint4*>(bptr + kb + 16 + bkoff);  // K = kb+16+bk..+7
        acc = __builtin_amdgcn_wmma_f32_16x16x32_bf16(
                  false, A.bf, false, B.bf, (short)0, acc, false, false);
    }
    // D layout: lane n = col, VGPR j -> row = row0 + hi*8 + j
    const int n = col0 + l15;
    const int rbase = row0 + hi * 8;
#pragma unroll
    for (int j = 0; j < 8; ++j) {
        int r = rbase + j;
        if (r < N_NODES) feat_src[(size_t)r * IN_F + n] = acc[j];
    }
}

// ---------------------------------------------------------------------------
// el/er: per (node, head) dot of feat_src slice with attn_l / attn_r
// ---------------------------------------------------------------------------
__global__ void k_elr(const float* __restrict__ feat_src,
                      const float* __restrict__ attn_l, const float* __restrict__ attn_r,
                      float* __restrict__ el, float* __restrict__ er) {
    int i = blockIdx.x * blockDim.x + threadIdx.x;
    if (i >= N_NODES * N_HEADS) return;
    int n = i >> 3, h = i & 7;
    const float* fs = feat_src + (size_t)n * IN_F + h * OUT_F;
    const float* al = attn_l + h * OUT_F;
    const float* ar = attn_r + h * OUT_F;
    float sl = 0.f, sr = 0.f;
#pragma unroll
    for (int d = 0; d < OUT_F; ++d) { float v = fs[d]; sl += v * al[d]; sr += v * ar[d]; }
    el[i] = sl; er[i] = sr;
}

// monotone float<->int mapping for atomicMax on floats
__device__ __forceinline__ int fflip(float x) {
    int i = __float_as_int(x);
    return (i >= 0) ? i : (i ^ 0x7fffffff);
}
__device__ __forceinline__ float funflip(int i) {
    return __int_as_float((i >= 0) ? i : (i ^ 0x7fffffff));
}

// ---------------------------------------------------------------------------
// per (edge, head): logits + segment-max via atomicMax(int)
// ---------------------------------------------------------------------------
__global__ void k_logits(const int* __restrict__ e_feat, const int* __restrict__ src,
                         const int* __restrict__ dst,
                         const float* __restrict__ el, const float* __restrict__ er,
                         const float* __restrict__ ee_t, const float* __restrict__ w_r,
                         float* __restrict__ logits, int* __restrict__ emax) {
    int i = blockIdx.x * blockDim.x + threadIdx.x;
    if (i >= N_EDGES * N_HEADS) return;
    int e = i >> 3, h = i & 7;
    int t = e_feat[e];
    int d = dst[e];
    float x = el[src[e] * N_HEADS + h] + er[d * N_HEADS + h] + ee_t[t * N_HEADS + h];
    x = (x > 0.f) ? x : 0.2f * x;         // leaky relu, NEG_SLOPE=0.2
    x *= w_r[t];                          // relation weight [8,1]
    logits[i] = x;
    atomicMax(&emax[d * N_HEADS + h], fflip(x));
}

// ---------------------------------------------------------------------------
// per (edge, head): ex = exp(logit - max), segment-sum via atomicAdd
// (exb may alias logits)
// ---------------------------------------------------------------------------
__global__ void k_exp(const int* __restrict__ dst, const float* __restrict__ logits,
                      const int* __restrict__ emax, float* __restrict__ exb,
                      float* __restrict__ denom) {
    int i = blockIdx.x * blockDim.x + threadIdx.x;
    if (i >= N_EDGES * N_HEADS) return;
    int e = i >> 3, h = i & 7;
    int dh = dst[e] * N_HEADS + h;
    float ex = __expf(logits[i] - funflip(emax[dh]));
    exb[i] = ex;
    atomicAdd(&denom[dh], ex);
}

// ---------------------------------------------------------------------------
// aggregation: one wave per edge. lane = feature d (0..31), heads unrolled.
// feat_src + rst both fit in 192MB L2 -> atomicAdd scatter stays on-chip.
// Also writes a = ex/denom to output.
// ---------------------------------------------------------------------------
__global__ void k_aggregate(const int* __restrict__ src, const int* __restrict__ dst,
                            const float* __restrict__ exb, const float* __restrict__ denom,
                            const float* __restrict__ feat_src,
                            float* __restrict__ rst, float* __restrict__ a_out) {
    int wave = (blockIdx.x * blockDim.x + threadIdx.x) >> 5;
    int lane = threadIdx.x & 31;
    if (wave >= N_EDGES) return;
    int s = src[wave], d = dst[wave];
    float a[N_HEADS];
#pragma unroll
    for (int h = 0; h < N_HEADS; ++h)
        a[h] = exb[(size_t)wave * N_HEADS + h] / denom[d * N_HEADS + h];
    if (lane < N_HEADS) a_out[(size_t)wave * N_HEADS + lane] = a[lane];
    const float* fsrc = feat_src + (size_t)s * IN_F;
    float* out = rst + (size_t)d * IN_F;
#pragma unroll
    for (int h = 0; h < N_HEADS; ++h)
        atomicAdd(&out[h * OUT_F + lane], a[h] * fsrc[h * OUT_F + lane]);
}

// ---------------------------------------------------------------------------
extern "C" void kernel_launch(void* const* d_in, const int* in_sizes, int n_in,
                              void* d_out, int out_size, void* d_ws, size_t ws_size,
                              hipStream_t stream) {
    const float* feat     = (const float*)d_in[0];
    const int*   e_feat   = (const int*)  d_in[1];
    const int*   src      = (const int*)  d_in[2];
    const int*   dst      = (const int*)  d_in[3];
    const float* fc_W     = (const float*)d_in[4];
    const float* edge_emb = (const float*)d_in[5];
    const float* w_r      = (const float*)d_in[6];
    const float* W1       = (const float*)d_in[7];
    const float* b1       = (const float*)d_in[8];
    const float* W2       = (const float*)d_in[9];
    const float* b2       = (const float*)d_in[10];
    const float* attn_l   = (const float*)d_in[11];
    const float* attn_r   = (const float*)d_in[12];
    const float* attn_e   = (const float*)d_in[13];

    float* rst   = (float*)d_out;                                   // [N, H, D]
    float* a_out = (float*)d_out + (size_t)N_NODES * N_HEADS * OUT_F; // [E, H, 1]

    // ---- workspace carve (byte offsets, 256B aligned) ----
    char* ws = (char*)d_ws;
    unsigned short* featb    = (unsigned short*)(ws);                      // 15,360,000 B
    unsigned short* fcwb     = (unsigned short*)(ws + 15360000);           //    131,072 B
    float*          feat_src = (float*)(ws + 15491072);                    // 30,720,000 B
    float*          el       = (float*)(ws + 46211072);                    //    960,000 B
    float*          er       = (float*)(ws + 47171072);                    //    960,000 B
    float*          logits   = (float*)(ws + 48131072);                    // 15,360,000 B
    int*            emax     = (int*)  (ws + 63491072);                    //    960,000 B
    float*          denom    = (float*)(ws + 64451072);                    //    960,000 B
    float*          ee_t     = (float*)(ws + 65411072);                    //        256 B

    const int NR = N_NODES * N_HEADS * OUT_F;   // 7,680,000
    const int ND = N_NODES * N_HEADS;           //   240,000
    const int EH = N_EDGES * N_HEADS;           // 3,840,000

    // init
    k_init<<<(NR + 255) / 256, 256, 0, stream>>>(rst, denom, emax, NR, ND);
    // bf16 conversions
    k_cvt_bf16<<<(N_NODES * IN_F + 255) / 256, 256, 0, stream>>>(feat, featb, N_NODES * IN_F);
    k_cvt_bf16<<<(IN_F * IN_F + 255) / 256, 256, 0, stream>>>(fc_W, fcwb, IN_F * IN_F);
    // per-type attention table
    k_typeprep<<<1, 128, 0, stream>>>(edge_emb, W1, b1, W2, b2, attn_e, ee_t);
    // WMMA GEMM: grid = (ceil(30000/32), 256/64)
    dim3 ggrid((N_NODES + 31) / 32, IN_F / 64);
    k_gemm_wmma<<<ggrid, 256, 0, stream>>>(featb, fcwb, feat_src);
    // el / er
    k_elr<<<(ND + 255) / 256, 256, 0, stream>>>(feat_src, attn_l, attn_r, el, er);
    // logits + segment max
    k_logits<<<(EH + 255) / 256, 256, 0, stream>>>(e_feat, src, dst, el, er, ee_t, w_r,
                                                   logits, emax);
    // exp + segment sum (exb aliases logits)
    k_exp<<<(EH + 255) / 256, 256, 0, stream>>>(dst, logits, emax, logits, denom);
    // aggregate (wave per edge) + emit a
    int agg_threads = N_EDGES * 32;
    k_aggregate<<<(agg_threads + 255) / 256, 256, 0, stream>>>(src, dst, logits, denom,
                                                               feat_src, rst, a_out);
}
